// MultiHeadAttention_83751862272628
// MI455X (gfx1250) — compile-verified, matrix operand order checked by reading, not run on hardware
//
#include <hip/hip_runtime.h>
#include <hip/hip_bf16.h>
#include <cstdint>

// ---------------------------------------------------------------------------
// MultiHeadAttention for MI455X (gfx1250, wave32, WMMA).
// B=8, S=1024, DM=512, H=8, DH=512 (HD = H*DH = 4096).
// v3: tile staging via global_load_async_to_lds_b128 (ASYNCcnt path),
//     double-buffered GEMM k-tiles, software-pipelined fragment loads so
//     WMMAs overlap ds_load_b128 latency instead of s_wait_dscnt 0 drains.
// ---------------------------------------------------------------------------

typedef __attribute__((ext_vector_type(16))) _Float16 v16h;
typedef __attribute__((ext_vector_type(8)))  _Float16 v8h;
typedef __attribute__((ext_vector_type(8)))  float    v8f;

#define BATCH 8
#define SEQ   1024
#define DMODEL 512
#define HEADS 8
#define DHEAD 512
#define HD    4096          // HEADS * DHEAD
#define ROWS  (BATCH * SEQ) // 8192

__device__ inline v8f wmma_f16(v16h a, v16h b, v8f c) {
  return __builtin_amdgcn_wmma_f32_16x16x32_f16(
      /*neg_a=*/false, a, /*neg_b=*/false, b,
      /*c_mod=*/(short)0, c, /*reuse_a=*/false, /*reuse_b=*/false);
}

__device__ inline v8f zero8() { v8f z = {}; return z; }

// Async 16B copy global -> LDS (GLOBAL_LOAD_ASYNC_TO_LDS_B128, ASYNCcnt).
// VDST = per-lane LDS byte address (low 32 bits of flat shared address).
__device__ inline void async_b128(uint32_t lds_addr, const void* gaddr) {
  asm volatile("global_load_async_to_lds_b128 %0, %1, off"
               :: "v"(lds_addr), "v"(gaddr)
               : "memory");
}
__device__ inline void wait_async0() {
  asm volatile("s_wait_asynccnt 0x0" ::: "memory");
}
__device__ inline uint32_t lds_off(const void* p) {
  return (uint32_t)(uintptr_t)p;
}

// Unified fragment load (A 16x32 row-major, or B 32x16 from [N][K] storage).
// Per ISA 7.12.2 the 16 halves of lane L are K [kb..kb+7] and [kb+16..kb+23]
// of row/col (L&15), kb=(L>>4)*8 -> two ds_load_b128 when K is contiguous.
__device__ inline v16h frag_ld(const _Float16* p, int ld, int lane) {
  const _Float16* pr = p + (lane & 15) * ld + ((lane >> 4) * 8);
  v8h lo = *reinterpret_cast<const v8h*>(pr);
  v8h hi = *reinterpret_cast<const v8h*>(pr + 16);
  return __builtin_shufflevector(lo, hi, 0, 1, 2, 3, 4, 5, 6, 7,
                                 8, 9, 10, 11, 12, 13, 14, 15);
}

// ---------------------------------------------------------------------------
// Elementwise f32 -> f16 convert (4 per thread)
// ---------------------------------------------------------------------------
__global__ __launch_bounds__(256) void cvt_f32_f16(const float* __restrict__ in,
                                                   _Float16* __restrict__ out,
                                                   int n) {
  int i = (blockIdx.x * 256 + threadIdx.x) * 4;
  if (i + 3 < n) {
    float4 v = *reinterpret_cast<const float4*>(in + i);
    out[i + 0] = (_Float16)v.x;
    out[i + 1] = (_Float16)v.y;
    out[i + 2] = (_Float16)v.z;
    out[i + 3] = (_Float16)v.w;
  } else {
    for (int j = i; j < n; ++j) out[j] = (_Float16)in[j];
  }
}

// ---------------------------------------------------------------------------
// Tiled transpose f32 -> f16: out[c][r] = in[r][c]  (R x C input)
// ---------------------------------------------------------------------------
__global__ __launch_bounds__(256) void transpose_f32_f16(
    const float* __restrict__ in, _Float16* __restrict__ out, int R, int C) {
  __shared__ float tile[32][33];
  const int bx = blockIdx.x * 32;
  const int by = blockIdx.y * 32;
  const int tx = threadIdx.x & 31;
  const int ty = threadIdx.x >> 5;
#pragma unroll
  for (int j = 0; j < 4; ++j)
    tile[ty + j * 8][tx] = in[(size_t)(by + ty + j * 8) * C + bx + tx];
  __syncthreads();
#pragma unroll
  for (int j = 0; j < 4; ++j)
    out[(size_t)(bx + ty + j * 8) * R + by + tx] =
        (_Float16)tile[tx][ty + j * 8];
}

// ---------------------------------------------------------------------------
// WMMA GEMM: C[M,N] = A[M,K] * BT[N,K]^T.  Block tile 128x256, K-step 32.
// 8 waves in 2x4, each wave 64x64 (16 wmma / k-step).
// Double-buffered LDS k-tiles staged with async-to-LDS copies.
// MODE: 0 = f32 out, 1 = f16 out, 2 = f16 out transposed to [b,h,d,s].
// ---------------------------------------------------------------------------
template <int MODE>
__global__ __launch_bounds__(256) void gemm_wmma(const _Float16* __restrict__ A,
                                                 const _Float16* __restrict__ BT,
                                                 void* __restrict__ Cv,
                                                 int M, int N, int K) {
  __shared__ __align__(16) _Float16 As[2][128 * 32];   // [buf][m][k]
  __shared__ __align__(16) _Float16 BsT[2][256 * 32];  // [buf][n][k]

  const int tid  = threadIdx.x;
  const int lane = tid & 31;
  const int w    = tid >> 5;
  const int m0   = blockIdx.y * 128;
  const int n0   = blockIdx.x * 256;
  const int wm   = (w & 1) * 64;
  const int wn   = (w >> 1) * 64;

  // Stage one k-tile into LDS buffer `buf` (async; 6 x b128 per thread).
  auto stage = [&](int k0, int buf) {
#pragma unroll
    for (int it = 0; it < 2; ++it) {  // A: 128 rows x 32 halves = 512 x 16B
      int i = tid + it * 256;
      int r = i >> 2, q = i & 3;
      async_b128(lds_off(&As[buf][r * 32 + q * 8]),
                 &A[(size_t)(m0 + r) * K + k0 + q * 8]);
    }
#pragma unroll
    for (int it = 0; it < 4; ++it) {  // BT: 256 n-rows x 32 k = 1024 x 16B
      int i = tid + it * 256;
      int r = i >> 2, q = i & 3;
      async_b128(lds_off(&BsT[buf][r * 32 + q * 8]),
                 &BT[(size_t)(n0 + r) * K + k0 + q * 8]);
    }
  };

  v8f acc[4][4];
#pragma unroll
  for (int im = 0; im < 4; ++im)
#pragma unroll
    for (int in = 0; in < 4; ++in) acc[im][in] = zero8();

  stage(0, 0);
  for (int k0 = 0; k0 < K; k0 += 32) {
    const int buf = (k0 >> 5) & 1;
    wait_async0();    // my copies for `buf` (and any in-flight) done
    __syncthreads();  // everyone's copies visible

    // Pipelined compute: preload B frags, rotate A frags across im steps.
    v16h bf[4];
#pragma unroll
    for (int in = 0; in < 4; ++in)
      bf[in] = frag_ld(&BsT[buf][(wn + in * 16) * 32], 32, lane);
    v16h af = frag_ld(&As[buf][wm * 32], 32, lane);
#pragma unroll
    for (int im = 0; im < 4; ++im) {
      v16h afn = af;
      if (im < 3) afn = frag_ld(&As[buf][(wm + (im + 1) * 16) * 32], 32, lane);
#pragma unroll
      for (int in = 0; in < 4; ++in)
        acc[im][in] = wmma_f16(af, bf[in], acc[im][in]);
      af = afn;
    }

    // Overlap: stage next tile into the other buffer while waves compute.
    if (k0 + 32 < K) stage(k0 + 32, buf ^ 1);
    if (k0 + 64 < K) {
      __builtin_prefetch(&A[(size_t)(m0 + (tid >> 1)) * K + k0 + 64], 0, 1);
      __builtin_prefetch(&BT[(size_t)(n0 + tid) * K + k0 + 64], 0, 1);
    }
  }

  // Epilogue. C/D layout: lane (L&15) = col, (L>>4)*8 + e = row.
  const int col = lane & 15;
  const int rb8 = (lane >> 4) * 8;
#pragma unroll
  for (int im = 0; im < 4; ++im)
#pragma unroll
    for (int in = 0; in < 4; ++in) {
      const int r0 = m0 + wm + im * 16 + rb8;
      const int c0 = n0 + wn + in * 16 + col;
      if constexpr (MODE == 0) {
#pragma unroll
        for (int e = 0; e < 8; ++e)
          ((float*)Cv)[(size_t)(r0 + e) * N + c0] = acc[im][in][e];
      } else if constexpr (MODE == 1) {
#pragma unroll
        for (int e = 0; e < 8; ++e)
          ((_Float16*)Cv)[(size_t)(r0 + e) * N + c0] = (_Float16)acc[im][in][e];
      } else {
        // Transposed store to [b, h, d, s]: row = b*SEQ+s, col = h*DHEAD+d.
        const int b = r0 >> 10, s0 = r0 & (SEQ - 1);
        const int h = c0 >> 9,  d  = c0 & (DHEAD - 1);
        v8h vals;
#pragma unroll
        for (int e = 0; e < 8; ++e) vals[e] = (_Float16)acc[im][in][e];
        _Float16* dst = (_Float16*)Cv +
                        (((size_t)b * HEADS + h) * DHEAD + d) * SEQ + s0;
        *reinterpret_cast<v8h*>(dst) = vals;
      }
    }
}

// ---------------------------------------------------------------------------
// Flash attention: one workgroup per (b, h, 64-query tile).
// Streams 16 blocks of 64 keys; online softmax; WMMA scores and P*V.
//   LDS: Q 64x512 | K 64x512 | V^T 512x64 (f16) | S 64x64 f32 | P 64x64 f16 |
//        m/l/rescale (64 f32 each)  (~218 KB of the 320 KB WGP LDS)
// ---------------------------------------------------------------------------
__global__ __launch_bounds__(256) void attn_flash(
    const _Float16* __restrict__ qh, const _Float16* __restrict__ kh,
    const _Float16* __restrict__ vhT, const unsigned char* __restrict__ mask,
    _Float16* __restrict__ ao) {
  extern __shared__ __align__(16) char smem_raw[];
  _Float16* Qs   = reinterpret_cast<_Float16*>(smem_raw);     // [64][512]
  _Float16* Ks   = Qs + 64 * 512;                             // [64][512]
  _Float16* VsT  = Ks + 64 * 512;                             // [512][64]
  float*    Ss   = reinterpret_cast<float*>(VsT + 512 * 64);  // [64][64]
  _Float16* Ps   = reinterpret_cast<_Float16*>(Ss + 64 * 64); // [64][64]
  float*    mrow = reinterpret_cast<float*>(Ps + 64 * 64);    // 64
  float*    lrow = mrow + 64;                                 // 64
  float*    rsc  = lrow + 64;                                 // 64

  const int tid  = threadIdx.x;
  const int lane = tid & 31;
  const int w    = tid >> 5;
  const int qt   = blockIdx.x;
  const int h    = blockIdx.y;
  const int b    = blockIdx.z;

  const size_t qbase  = ((size_t)b * SEQ + qt * 64) * HD + (size_t)h * DHEAD;
  const size_t vtbase = ((size_t)b * HEADS + h) * DHEAD * SEQ;  // [d][s]

  // Q tile via async copies (drained by the first wait in the j-loop).
#pragma unroll
  for (int it = 0; it < 16; ++it) {
    int i = tid + it * 256;
    int r = i >> 6, q = i & 63;
    async_b128(lds_off(&Qs[r * 512 + q * 8]),
               &qh[qbase + (size_t)r * HD + q * 8]);
  }
  if (tid < 64) { mrow[tid] = -INFINITY; lrow[tid] = 0.0f; }

  const int rb = (w & 3);   // output rows rb*16 .. rb*16+15
  const int ch = (w >> 2);  // output cols ch*256 .. +255
  v8f acc[16];
#pragma unroll
  for (int t = 0; t < 16; ++t) acc[t] = zero8();

  const float sm_scale = 0.044194173824159216f;  // 1/sqrt(512)

  for (int j = 0; j < 16; ++j) {
    __syncthreads();  // previous P*V done before K/V overwrite
    const size_t kbase = ((size_t)b * SEQ + j * 64) * HD + (size_t)h * DHEAD;
#pragma unroll
    for (int it = 0; it < 16; ++it) {  // K block 64x512 (async b128)
      int i = tid + it * 256;
      int r = i >> 6, q = i & 63;
      async_b128(lds_off(&Ks[r * 512 + q * 8]),
                 &kh[kbase + (size_t)r * HD + q * 8]);
    }
#pragma unroll
    for (int it = 0; it < 16; ++it) {  // V^T block 512x64 (async b128)
      int i = tid + it * 256;
      int r = i >> 3, q = i & 7;
      async_b128(lds_off(&VsT[r * 64 + q * 8]),
                 &vhT[vtbase + (size_t)r * SEQ + j * 64 + q * 8]);
    }
    if (j + 1 < 16) {  // prefetch next K/V block
      const size_t kn = kbase + (size_t)64 * HD;
      __builtin_prefetch(&kh[kn + (size_t)(tid >> 2) * HD + (tid & 3) * 128], 0, 1);
      __builtin_prefetch(&vhT[vtbase + (size_t)(tid * 2) * SEQ + (j + 1) * 64], 0, 1);
    }
    wait_async0();
    __syncthreads();

    // Scores S = Q * K^T (4x4 grid of 16x16 tiles; 2 tiles per wave),
    // fragment loads pipelined one k-step ahead.
#pragma unroll
    for (int tt = 0; tt < 2; ++tt) {
      const int t  = w * 2 + tt;
      const int tr = t >> 2, tc = t & 3;
      v8f s = zero8();
      v16h a_c = frag_ld(&Qs[(tr * 16) * 512], 512, lane);
      v16h b_c = frag_ld(&Ks[(tc * 16) * 512], 512, lane);
#pragma unroll
      for (int ks = 0; ks < 16; ++ks) {
        v16h a_n = a_c, b_n = b_c;
        if (ks < 15) {
          a_n = frag_ld(&Qs[(tr * 16) * 512 + (ks + 1) * 32], 512, lane);
          b_n = frag_ld(&Ks[(tc * 16) * 512 + (ks + 1) * 32], 512, lane);
        }
        s = wmma_f16(a_c, b_c, s);
        a_c = a_n;
        b_c = b_n;
      }
      const int col   = tc * 16 + (lane & 15);
      const int rbase = tr * 16 + (lane >> 4) * 8;
#pragma unroll
      for (int e = 0; e < 8; ++e) Ss[(rbase + e) * 64 + col] = s[e] * sm_scale;
    }
    __syncthreads();

    // Online softmax update: one thread per query row.
    if (tid < 64) {
      const int r = tid;
      const unsigned char* mp =
          mask + (((size_t)b * SEQ) + qt * 64 + r) * SEQ + j * 64;
      const float mold = mrow[r];
      float mnew = mold;
      for (int k = 0; k < 64; ++k) {
        float s = mp[k] ? -1e9f : Ss[r * 64 + k];
        mnew = fmaxf(mnew, s);
      }
      const float scale = __expf(mold - mnew);
      float sum = 0.0f;
      for (int k = 0; k < 64; ++k) {
        float s = mp[k] ? -1e9f : Ss[r * 64 + k];
        float p = __expf(s - mnew);
        sum += p;
        Ps[r * 64 + k] = (_Float16)p;
      }
      mrow[r] = mnew;
      lrow[r] = lrow[r] * scale + sum;
      rsc[r]  = scale;
    }
    __syncthreads();

    // Rescale accumulators by exp(m_old - m_new) per row.
    {
      const int rbase = rb * 16 + (lane >> 4) * 8;
#pragma unroll
      for (int e = 0; e < 8; ++e) {
        const float f = rsc[rbase + e];
#pragma unroll
        for (int t = 0; t < 16; ++t) acc[t][e] *= f;
      }
    }

    // O += P(64x64) * V(64x512); V^T fragments pipelined one tile ahead.
#pragma unroll
    for (int kk = 0; kk < 2; ++kk) {
      v16h a  = frag_ld(&Ps[(rb * 16) * 64 + kk * 32], 64, lane);
      v16h bc = frag_ld(&VsT[(ch * 256) * 64 + kk * 32], 64, lane);
#pragma unroll
      for (int t = 0; t < 16; ++t) {
        v16h bn = bc;
        if (t < 15)
          bn = frag_ld(&VsT[(ch * 256 + (t + 1) * 16) * 64 + kk * 32], 64, lane);
        acc[t] = wmma_f16(a, bc, acc[t]);
        bc = bn;
      }
    }
  }
  __syncthreads();

  // Normalize by row sum, store f16 to concat-head layout [b, s, h*512+d].
  {
    const int rbase = rb * 16 + (lane >> 4) * 8;
    const int col   = lane & 15;
#pragma unroll
    for (int e = 0; e < 8; ++e) {
      const float linv = 1.0f / lrow[rbase + e];
      const int   grow = qt * 64 + rbase + e;
      const size_t ob  = ((size_t)b * SEQ + grow) * HD + (size_t)h * DHEAD +
                         ch * 256 + col;
#pragma unroll
      for (int t = 0; t < 16; ++t)
        ao[ob + t * 16] = (_Float16)(acc[t][e] * linv);
    }
  }
}

// ---------------------------------------------------------------------------
// LayerNorm over rows of 512 (one block per row, 256 threads x 2 elems)
// ---------------------------------------------------------------------------
__global__ __launch_bounds__(256) void layernorm_rows(
    const float* __restrict__ x, const float* __restrict__ gamma,
    const float* __restrict__ beta, float* __restrict__ out) {
  __shared__ float red[256];
  const int row = blockIdx.x;
  const int t   = threadIdx.x;
  const float* xr = x + (size_t)row * DMODEL;
  const float a0 = xr[t];
  const float a1 = xr[t + 256];

  red[t] = a0 + a1;
  __syncthreads();
#pragma unroll
  for (int off = 128; off > 0; off >>= 1) {
    if (t < off) red[t] += red[t + off];
    __syncthreads();
  }
  const float mu = red[0] * (1.0f / DMODEL);
  __syncthreads();

  const float d0 = a0 - mu, d1 = a1 - mu;
  red[t] = d0 * d0 + d1 * d1;
  __syncthreads();
#pragma unroll
  for (int off = 128; off > 0; off >>= 1) {
    if (t < off) red[t] += red[t + off];
    __syncthreads();
  }
  const float var  = red[0] * (1.0f / DMODEL);
  const float rstd = rsqrtf(var + 1e-5f);

  out[(size_t)row * DMODEL + t]       = d0 * rstd * gamma[t]       + beta[t];
  out[(size_t)row * DMODEL + t + 256] = d1 * rstd * gamma[t + 256] + beta[t + 256];
}

// ---------------------------------------------------------------------------
// Host launcher
// ---------------------------------------------------------------------------
extern "C" void kernel_launch(void* const* d_in, const int* in_sizes, int n_in,
                              void* d_out, int out_size, void* d_ws,
                              size_t ws_size, hipStream_t stream) {
  (void)in_sizes; (void)n_in; (void)out_size; (void)ws_size;

  const float* q     = (const float*)d_in[0];
  const float* k     = (const float*)d_in[1];
  const float* v     = (const float*)d_in[2];
  const float* Wq    = (const float*)d_in[3];
  const float* Wk    = (const float*)d_in[4];
  const float* Wv    = (const float*)d_in[5];
  const float* Wo    = (const float*)d_in[6];
  const float* gamma = (const float*)d_in[7];
  const float* beta  = (const float*)d_in[8];
  const unsigned char* mask = (const unsigned char*)d_in[9];
  float* out = (float*)d_out;

  char* ws = (char*)d_ws;
  size_t off = 0;
  auto carve = [&](size_t bytes) -> void* {
    off = (off + 255) & ~(size_t)255;
    void* p = ws + off;
    off += bytes;
    return p;
  };
  const size_t nQKV = (size_t)ROWS * DMODEL;
  const size_t nW   = (size_t)DMODEL * HD;
  const size_t nH   = (size_t)ROWS * HD;

  _Float16* qf16 = (_Float16*)carve(nQKV * 2);
  _Float16* kf16 = (_Float16*)carve(nQKV * 2);
  _Float16* vf16 = (_Float16*)carve(nQKV * 2);
  _Float16* WqT  = (_Float16*)carve(nW * 2);  // [HD][DMODEL]
  _Float16* WkT  = (_Float16*)carve(nW * 2);
  _Float16* WvT  = (_Float16*)carve(nW * 2);
  _Float16* WoT  = (_Float16*)carve(nW * 2);  // [DMODEL][HD]
  _Float16* qhB  = (_Float16*)carve(nH * 2);  // [b*S+s][h*DH+d]
  _Float16* khB  = (_Float16*)carve(nH * 2);
  _Float16* vhT  = (_Float16*)carve(nH * 2);  // [b][h][d][s]
  _Float16* aoB  = (_Float16*)carve(nH * 2);
  float*    proj = (float*)carve((size_t)ROWS * DMODEL * 4);

  auto cvt = [&](const float* src, _Float16* dst, size_t n) {
    int blocks = (int)((n / 4 + 255) / 256);
    cvt_f32_f16<<<blocks, 256, 0, stream>>>(src, dst, (int)n);
  };
  cvt(q, qf16, nQKV);
  cvt(k, kf16, nQKV);
  cvt(v, vf16, nQKV);

  transpose_f32_f16<<<dim3(HD / 32, DMODEL / 32), 256, 0, stream>>>(Wq, WqT, DMODEL, HD);
  transpose_f32_f16<<<dim3(HD / 32, DMODEL / 32), 256, 0, stream>>>(Wk, WkT, DMODEL, HD);
  transpose_f32_f16<<<dim3(HD / 32, DMODEL / 32), 256, 0, stream>>>(Wv, WvT, DMODEL, HD);
  transpose_f32_f16<<<dim3(DMODEL / 32, HD / 32), 256, 0, stream>>>(Wo, WoT, HD, DMODEL);

  dim3 gProj(HD / 256, ROWS / 128);
  gemm_wmma<1><<<gProj, 256, 0, stream>>>(qf16, WqT, qhB, ROWS, HD, DMODEL);
  gemm_wmma<1><<<gProj, 256, 0, stream>>>(kf16, WkT, khB, ROWS, HD, DMODEL);
  gemm_wmma<2><<<gProj, 256, 0, stream>>>(vf16, WvT, vhT, ROWS, HD, DMODEL);

  const size_t shmem = (size_t)(3 * 64 * 512) * 2
                     + (size_t)(64 * 64) * 4
                     + (size_t)(64 * 64) * 2
                     + (size_t)(3 * 64) * 4;
  dim3 gAttn(SEQ / 64, HEADS, BATCH);
  attn_flash<<<gAttn, 256, shmem, stream>>>(qhB, khB, vhT, mask, aoB);

  dim3 gOut(DMODEL / 256, ROWS / 128);
  gemm_wmma<0><<<gOut, 256, 0, stream>>>(aoB, WoT, proj, ROWS, DMODEL, HD);

  layernorm_rows<<<ROWS, 256, 0, stream>>>(proj, gamma, beta, out);
}
